// BertLayerWithMemory_82068235092831
// MI455X (gfx1250) — matrix operand order repaired and checked by reading, not compiled, not run on hardware
//
#include <hip/hip_runtime.h>

// ---------------------------------------------------------------------------
// MART BertLayerWithMemory forward on gfx1250 (MI455X), bf16 WMMA everywhere.
//   - dense projections: double-buffered tiled GEMM, LDS tiles filled by the
//     Tensor Data Mover (tensor_load_to_lds + s_wait_tensorcnt), compute via
//     v_wmma_f32_16x16x32_bf16
//   - all three attentions: flash-style online-softmax, QK^T and PV via WMMA,
//     P-fragment relayout + V transpose through LDS
//   - fused epilogues (bias/gelu/tanh/sigmoid/extra-add, bf16 or f32 out)
// ---------------------------------------------------------------------------

typedef __attribute__((ext_vector_type(16))) __bf16 v16bf;
typedef __attribute__((ext_vector_type(8)))  float  v8f;
typedef __attribute__((ext_vector_type(4)))  unsigned int v4u;
typedef __attribute__((ext_vector_type(8)))  int v8i;
typedef __attribute__((ext_vector_type(4)))  int v4i;

#define DMODEL 1024
#define NHEAD  16
#define DHEAD  64
#define NB     4
#define LSEQ   1024
#define MMEM   128

#if __has_builtin(__builtin_amdgcn_tensor_load_to_lds) && \
    __has_builtin(__builtin_amdgcn_s_wait_tensorcnt)
#define USE_TDM 1
#else
#define USE_TDM 0
#endif

__device__ inline unsigned short f2bf(float x) {
  union { float f; unsigned u; } v; v.f = x;
  unsigned r = v.u + 0x7FFFu + ((v.u >> 16) & 1u);   // round-to-nearest-even
  return (unsigned short)(r >> 16);
}

union FragAB { v16bf v; unsigned int u[8]; };

#if USE_TDM
// Issue one 2-D TDM tile load: global (strided rows) -> LDS (contiguous).
// tile0 = row length (elems), tile1 = #rows, stride = row stride (elems),
// all element units are 2 bytes (bf16).
__device__ inline void tdm_tile_load(const unsigned short* g, unsigned ldsoff,
                                     int tile0, int tile1, unsigned dim0,
                                     unsigned dim1, unsigned long long stride) {
  unsigned long long ga = (unsigned long long)(const void*)g;
  v4u g0;
  g0[0] = 1u;                                         // count=1, user mode
  g0[1] = ldsoff;                                     // lds_addr (bytes)
  g0[2] = (unsigned)(ga & 0xFFFFFFFFull);             // global_addr[31:0]
  g0[3] = (unsigned)((ga >> 32) & 0x1FFFFFFull) | 0x80000000u;  // [56:32]|type=2
  v8i g1;
  g1[0] = 0x00010000;                                 // data_size=1 (2 bytes)
  g1[1] = (int)((dim0 & 0xFFFFu) << 16);              // tensor_dim0[15:0]
  g1[2] = (int)(((dim0 >> 16) & 0xFFFFu) | ((dim1 & 0xFFFFu) << 16));
  g1[3] = (int)(((dim1 >> 16) & 0xFFFFu) | ((unsigned)tile0 << 16));  // tile_dim0
  g1[4] = (int)((unsigned)tile1 & 0xFFFFu);           // tile_dim1, tile_dim2=0
  g1[5] = (int)((unsigned)(stride & 0xFFFFFFFFull));  // tensor_dim0_stride lo
  g1[6] = (int)((unsigned)((stride >> 32) & 0xFFFFull));  // stride hi16
  g1[7] = 0;
  v4i z4 = {0, 0, 0, 0};                              // groups 2/3: 2-D tensor
  v8i z8 = {0, 0, 0, 0, 0, 0, 0, 0};
  __builtin_amdgcn_tensor_load_to_lds(g0, g1, z4, z4, z8, 0);
}
#endif

// ---------------------------------------------------------------------------
// Tiled bf16 GEMM: C[M,N] = A[M,K] * B[K,N]  (row-major, M%64==0, N%64==0,
// K%32==0).  Block = 128 threads = 4 waves; each wave owns a 32x32 tile
// (2x2 WMMA 16x16 tiles), block tile = 64x64.  LDS tiles are double-buffered
// and filled by TDM (one A quarter-tile + one B quarter-tile per wave).
// FLAGS: 1=+bias  2=gelu  4=tanh  8=sigmoid  16=+extra(f32)  32=f32 out
//        64=bf16 out
// ---------------------------------------------------------------------------
template <int FLAGS>
__global__ __launch_bounds__(128) void gemm_bf16(
    const unsigned short* __restrict__ A, const unsigned short* __restrict__ B,
    const float* __restrict__ bias, const float* __restrict__ extra,
    float* __restrict__ outF, unsigned short* __restrict__ outB,
    int M, int N, int K) {
  __shared__ unsigned short As[2][64][32];   // A tiles, row-major
  __shared__ unsigned short Bs[2][32][64];   // B tiles, row-major [k][n]
  const int tid  = threadIdx.x;
  const int lane = tid & 31, wave = tid >> 5;
  const int half = lane >> 4, l15 = lane & 15;
  const int wm = wave >> 1, wn = wave & 1;
  const int row0 = blockIdx.y * 64, col0 = blockIdx.x * 64;

  auto stage = [&](int buf, int kk) {
#if USE_TDM
    // per-wave DMA: 16 rows of A (16x32), 8 rows of B (8x64)
    tdm_tile_load(&A[(size_t)(row0 + wave * 16) * K + kk],
                  (unsigned)(unsigned long long)(const void*)&As[buf][wave * 16][0],
                  32, 16, (unsigned)K, (unsigned)M, (unsigned long long)K);
    tdm_tile_load(&B[(size_t)(kk + wave * 8) * N + col0],
                  (unsigned)(unsigned long long)(const void*)&Bs[buf][wave * 8][0],
                  64, 8, (unsigned)N, (unsigned)K, (unsigned long long)N);
#else
    for (int i = 0; i < 8; ++i) {          // A: 1024 dwords
      int idx = tid + i * 128;
      int r = idx >> 4, c = (idx & 15) << 1;
      *(unsigned int*)&As[buf][r][c] =
          *(const unsigned int*)&A[(size_t)(row0 + r) * K + kk + c];
    }
    for (int i = 0; i < 8; ++i) {          // B: 1024 dwords, row-major copy
      int idx = tid + i * 128;
      int kr = idx >> 5, c2 = (idx & 31) << 1;
      *(unsigned int*)&Bs[buf][kr][c2] =
          *(const unsigned int*)&B[(size_t)(kk + kr) * N + col0 + c2];
    }
#endif
  };
  auto wait_tiles = [&]() {
#if USE_TDM
    __builtin_amdgcn_s_wait_tensorcnt(0);
#endif
    __syncthreads();
  };

  v8f acc[2][2] = {};
  stage(0, 0);
  wait_tiles();
  for (int kk = 0; kk < K; kk += 32) {
    const int buf = (kk >> 5) & 1;
    if (kk + 32 < K) stage(buf ^ 1, kk + 32);   // DMA next tile during compute

    FragAB af[2], bfb[2];
    for (int mi = 0; mi < 2; ++mi) {
      int r = wm * 32 + mi * 16 + l15;
      for (int j = 0; j < 8; ++j) {
        int k0 = 2 * j + 16 * half;  // A layout
        af[mi].u[j] = *(const unsigned int*)&As[buf][r][k0];
      }
    }
    for (int ni = 0; ni < 2; ++ni) {
      int c = wn * 32 + ni * 16 + l15;
      for (int j = 0; j < 8; ++j) {
        int kr = 2 * j + 16 * half;                               // B layout
        unsigned lo = Bs[buf][kr][c];
        unsigned hi = Bs[buf][kr + 1][c];
        bfb[ni].u[j] = lo | (hi << 16);
      }
    }
    for (int mi = 0; mi < 2; ++mi)
      for (int ni = 0; ni < 2; ++ni)
        acc[mi][ni] = __builtin_amdgcn_wmma_f32_16x16x32_bf16(
            false, af[mi].v, false, bfb[ni].v, (short)0, acc[mi][ni], false,
            false);
    wait_tiles();
  }
  // epilogue: C layout = VGPR r -> row (r + 8*half), col = l15
  for (int mi = 0; mi < 2; ++mi)
    for (int ni = 0; ni < 2; ++ni)
      for (int r = 0; r < 8; ++r) {
        int grow = row0 + wm * 32 + mi * 16 + r + 8 * half;
        int gcol = col0 + wn * 32 + ni * 16 + l15;
        float v = acc[mi][ni][r];
        if (FLAGS & 1)  v += bias[gcol];
        if (FLAGS & 16) v += extra[(size_t)grow * N + gcol];
        if (FLAGS & 2)  v = 0.5f * v * (1.0f + erff(v * 0.70710678118654752f));
        if (FLAGS & 4)  v = tanhf(v);
        if (FLAGS & 8)  v = 1.0f / (1.0f + expf(-v));
        size_t o = (size_t)grow * N + gcol;
        if (FLAGS & 32) outF[o] = v;
        if (FLAGS & 64) outB[o] = f2bf(v);
      }
}

// ---------------------------------------------------------------------------
// Flash attention (one wave per block, 16 q-rows, key tiles of 32).
// Q,K,V,Out: bf16 (N, Lrows, DMODEL) with head h at columns h*64..h*64+63.
// Mask: additive -10000 exactly like the reference.
// ---------------------------------------------------------------------------
__device__ inline float mask_add(const float* __restrict__ mv, int n, int Lk,
                                 int k, int q, int memlen, int causal) {
  float m = mv[(size_t)n * Lk + k];
  bool ok = (k < memlen + 512);
  if (!ok && causal) ok = (q >= 512) && ((k - (memlen + 512)) <= (q - 512));
  if (!ok) m = 0.0f;
  return (1.0f - m) * -10000.0f;
}

__global__ __launch_bounds__(32) void flash_attn(
    const unsigned short* __restrict__ Q, const unsigned short* __restrict__ Kb,
    const unsigned short* __restrict__ Vb, const float* __restrict__ maskvec,
    unsigned short* __restrict__ Out, int Lq, int Lk, int memlen, int causal) {
  __shared__ unsigned short Pl[16][32];   // exp(scores) tile as bf16 (A relayout)
  __shared__ unsigned short Vt[64][32];   // V tile transposed: Vt[dcol][key]
  const int lane = threadIdx.x & 31;
  const int half = lane >> 4, l15 = lane & 15;
  const int n = blockIdx.z, h = blockIdx.y, qbase = blockIdx.x * 16;
  const int Dm = DMODEL;

  const unsigned short* qp = Q + ((size_t)(n * Lq + qbase) * Dm + h * DHEAD);
  FragAB qf[2];
  for (int dj = 0; dj < 2; ++dj)
    for (int j = 0; j < 8; ++j) {
      int k0 = 2 * j + 16 * half + 32 * dj;
      qf[dj].u[j] = *(const unsigned int*)(qp + (size_t)l15 * Dm + k0);
    }

  v8f acc[4] = {};
  float mrow[8], lrow[8];
  for (int r = 0; r < 8; ++r) { mrow[r] = -1e30f; lrow[r] = 0.0f; }
  const float scale = 0.125f;   // 1/sqrt(64)

  for (int kt = 0; kt < Lk; kt += 32) {
    if (kt + 32 < Lk)            // CDNA5 global_prefetch for next key tile
      __builtin_prefetch(
          Kb + ((size_t)(n * Lk + kt + 32 + lane) * Dm + h * DHEAD), 0, 1);
    // ---- scores: two 16x16 tiles via WMMA over dh (=2 x k32) -------------
    v8f st[2] = {};
    for (int sb = 0; sb < 2; ++sb) {
      int key = kt + sb * 16 + l15;
      const unsigned short* kp = Kb + ((size_t)(n * Lk + key) * Dm + h * DHEAD);
      for (int dj = 0; dj < 2; ++dj) {
        FragAB bfr;
        for (int j = 0; j < 8; ++j)
          bfr.u[j] = *(const unsigned int*)(kp + 2 * j + 16 * half + 32 * dj);
        st[sb] = __builtin_amdgcn_wmma_f32_16x16x32_bf16(
            false, qf[dj].v, false, bfr.v, (short)0, st[sb], false, false);
      }
    }
    // ---- mask + online softmax ------------------------------------------
    float s0[8], s1[8];
    const int key0 = kt + l15, key1 = kt + 16 + l15;
    if (kt + 31 < memlen + 512) {
      // structural mask is q-independent for this whole tile: hoist it
      float madd0 = (1.0f - maskvec[(size_t)n * Lk + key0]) * -10000.0f;
      float madd1 = (1.0f - maskvec[(size_t)n * Lk + key1]) * -10000.0f;
      for (int r = 0; r < 8; ++r) {
        s0[r] = st[0][r] * scale + madd0;
        s1[r] = st[1][r] * scale + madd1;
      }
    } else {
      for (int r = 0; r < 8; ++r) {
        int q = qbase + r + 8 * half;
        s0[r] = st[0][r] * scale + mask_add(maskvec, n, Lk, key0, q, memlen, causal);
        s1[r] = st[1][r] * scale + mask_add(maskvec, n, Lk, key1, q, memlen, causal);
      }
    }
    for (int r = 0; r < 8; ++r) {
      float v = fmaxf(s0[r], s1[r]);
      for (int off = 8; off; off >>= 1) v = fmaxf(v, __shfl_xor(v, off, 32));
      float mn = fmaxf(mrow[r], v);
      float a  = expf(mrow[r] - mn);
      float p0 = expf(s0[r] - mn), p1 = expf(s1[r] - mn);
      float ps = p0 + p1;
      for (int off = 8; off; off >>= 1) ps += __shfl_xor(ps, off, 32);
      lrow[r] = lrow[r] * a + ps;
      mrow[r] = mn;
      s0[r] = p0; s1[r] = p1;
      for (int nb = 0; nb < 4; ++nb) acc[nb][r] *= a;
    }
    // ---- relayout P (C->A) through LDS, stage V transposed ---------------
    __syncthreads();
    for (int r = 0; r < 8; ++r) {
      int row = r + 8 * half;
      Pl[row][l15]      = f2bf(s0[r]);
      Pl[row][16 + l15] = f2bf(s1[r]);
    }
    {
      const unsigned short* vp =
          Vb + ((size_t)(n * Lk + kt + lane) * Dm + h * DHEAD);
      for (int t = 0; t < 16; ++t) {
        unsigned int u = *(const unsigned int*)(vp + 2 * t);
        Vt[2 * t][lane]     = (unsigned short)(u & 0xFFFFu);
        Vt[2 * t + 1][lane] = (unsigned short)(u >> 16);
      }
    }
    __syncthreads();
    // ---- PV accumulate ----------------------------------------------------
    FragAB pf;
    for (int j = 0; j < 8; ++j) {
      int k0 = 2 * j + 16 * half;
      pf.u[j] = *(const unsigned int*)&Pl[l15][k0];
    }
    for (int nb = 0; nb < 4; ++nb) {
      FragAB bfr;
      for (int j = 0; j < 8; ++j)
        bfr.u[j] = *(const unsigned int*)&Vt[nb * 16 + l15][2 * j + 16 * half];
      acc[nb] = __builtin_amdgcn_wmma_f32_16x16x32_bf16(
          false, pf.v, false, bfr.v, (short)0, acc[nb], false, false);
    }
  }
  // ---- write context (bf16) -----------------------------------------------
  unsigned short* op = Out + ((size_t)(n * Lq + qbase) * Dm + h * DHEAD);
  for (int nb = 0; nb < 4; ++nb)
    for (int r = 0; r < 8; ++r) {
      int row = r + 8 * half, col = nb * 16 + l15;
      op[(size_t)row * Dm + col] = f2bf(acc[nb][r] / fmaxf(lrow[r], 1e-30f));
    }
}

// ---------------------------------------------------------------------------
// Elementwise / normalization helpers
// ---------------------------------------------------------------------------
__global__ void f32_to_bf16_k(const float* __restrict__ in,
                              unsigned short* __restrict__ out, int n) {
  for (int i = blockIdx.x * blockDim.x + threadIdx.x; i < n;
       i += gridDim.x * blockDim.x)
    out[i] = f2bf(in[i]);
}

__global__ __launch_bounds__(256) void add_ln_k(
    const float* __restrict__ x, const float* __restrict__ resid,
    const float* __restrict__ g, const float* __restrict__ b,
    float* __restrict__ outF, unsigned short* __restrict__ outB) {
  __shared__ float red[256];
  const int row = blockIdx.x, tid = threadIdx.x;
  float v[4];
  for (int i = 0; i < 4; ++i) {
    size_t o = (size_t)row * DMODEL + tid + 256 * i;
    v[i] = x[o] + resid[o];
  }
  float s = v[0] + v[1] + v[2] + v[3];
  red[tid] = s; __syncthreads();
  for (int o = 128; o; o >>= 1) { if (tid < o) red[tid] += red[tid + o]; __syncthreads(); }
  float mean = red[0] / DMODEL;
  __syncthreads();
  float sq = 0.0f;
  for (int i = 0; i < 4; ++i) { float d = v[i] - mean; sq += d * d; }
  red[tid] = sq; __syncthreads();
  for (int o = 128; o; o >>= 1) { if (tid < o) red[tid] += red[tid + o]; __syncthreads(); }
  float inv = rsqrtf(red[0] / DMODEL + 1e-12f);
  for (int i = 0; i < 4; ++i) {
    int col = tid + 256 * i;
    size_t o = (size_t)row * DMODEL + col;
    float y = g[col] * (v[i] - mean) * inv + b[col];
    outF[o] = y;
    if (outB) outB[o] = f2bf(y);
  }
}

__global__ void gate_update_k(const float* __restrict__ c,
                              const float* __restrict__ z,
                              const float* __restrict__ pm,
                              float* __restrict__ out, int n) {
  for (int i = blockIdx.x * blockDim.x + threadIdx.x; i < n;
       i += gridDim.x * blockDim.x)
    out[i] = (1.0f - z[i]) * c[i] + z[i] * pm[i];
}

__global__ void concat_rows_k(const unsigned short* __restrict__ pm,
                              const unsigned short* __restrict__ inter,
                              unsigned short* __restrict__ out, int n) {
  const int rowlen = (MMEM + LSEQ) * DMODEL;
  for (int i = blockIdx.x * blockDim.x + threadIdx.x; i < n;
       i += gridDim.x * blockDim.x) {
    int nb  = i / rowlen;
    int rem = i - nb * rowlen;
    int row = rem / DMODEL, col = rem - row * DMODEL;
    out[i] = (row < MMEM)
                 ? pm[((size_t)nb * MMEM + row) * DMODEL + col]
                 : inter[((size_t)nb * LSEQ + (row - MMEM)) * DMODEL + col];
  }
}

__global__ void build_ma_mask_k(const float* __restrict__ am,
                                float* __restrict__ out, int n) {
  const int Lk = MMEM + LSEQ;
  for (int i = blockIdx.x * blockDim.x + threadIdx.x; i < n;
       i += gridDim.x * blockDim.x) {
    int nb = i / Lk, k = i - nb * Lk;
    out[i] = (k < MMEM) ? 1.0f : am[(size_t)nb * LSEQ + (k - MMEM)];
  }
}

// ---------------------------------------------------------------------------
// Host orchestration
// ---------------------------------------------------------------------------
extern "C" void kernel_launch(void* const* d_in, const int* in_sizes, int n_in,
                              void* d_out, int out_size, void* d_ws,
                              size_t ws_size, hipStream_t stream) {
  (void)in_sizes; (void)n_in; (void)out_size; (void)ws_size;
  const float* prev_m = (const float*)d_in[0];
  const float* hidden = (const float*)d_in[1];
  const float* amask  = (const float*)d_in[2];
  const float* W_attq = (const float*)d_in[3],  *b_attq = (const float*)d_in[4];
  const float* W_attk = (const float*)d_in[5],  *b_attk = (const float*)d_in[6];
  const float* W_attv = (const float*)d_in[7],  *b_attv = (const float*)d_in[8];
  const float* W_atto = (const float*)d_in[9],  *b_atto = (const float*)d_in[10];
  const float* ln1_g  = (const float*)d_in[11], *ln1_b  = (const float*)d_in[12];
  const float* W_int  = (const float*)d_in[13], *b_int  = (const float*)d_in[14];
  const float* W_muq  = (const float*)d_in[15], *b_muq  = (const float*)d_in[16];
  const float* W_muk  = (const float*)d_in[17], *b_muk  = (const float*)d_in[18];
  const float* W_muv  = (const float*)d_in[19], *b_muv  = (const float*)d_in[20];
  const float* W_mc   = (const float*)d_in[21];
  const float* W_sc   = (const float*)d_in[22], *b_sc   = (const float*)d_in[23];
  const float* W_mz   = (const float*)d_in[24];
  const float* W_sz   = (const float*)d_in[25], *b_sz   = (const float*)d_in[26];
  const float* W_maq  = (const float*)d_in[27], *b_maq  = (const float*)d_in[28];
  const float* W_mak  = (const float*)d_in[29], *b_mak  = (const float*)d_in[30];
  const float* W_mav  = (const float*)d_in[31], *b_mav  = (const float*)d_in[32];
  const float* W_mp   = (const float*)d_in[33];
  const float* W_out  = (const float*)d_in[34], *b_out  = (const float*)d_in[35];
  const float* ln2_g  = (const float*)d_in[36], *ln2_b  = (const float*)d_in[37];

  const int D = DMODEL;
  const int NL = NB * LSEQ;              // 4096
  const int NM = NB * MMEM;              // 512
  const int NC = NB * (MMEM + LSEQ);     // 4608
  const size_t DD = (size_t)D * D;

  char* base = (char*)d_ws; size_t off = 0;
  auto alloc = [&](size_t bytes) -> void* {
    off = (off + 255) & ~(size_t)255;
    void* p = base + off; off += bytes; return p;
  };
  typedef unsigned short u16;
  u16* wq  = (u16*)alloc(DD*2); u16* wk  = (u16*)alloc(DD*2); u16* wv  = (u16*)alloc(DD*2);
  u16* wo  = (u16*)alloc(DD*2); u16* wi  = (u16*)alloc(DD*2);
  u16* wmq = (u16*)alloc(DD*2); u16* wmk = (u16*)alloc(DD*2); u16* wmv = (u16*)alloc(DD*2);
  u16* wmc = (u16*)alloc(DD*2); u16* wsc = (u16*)alloc(DD*2);
  u16* wmz = (u16*)alloc(DD*2); u16* wsz = (u16*)alloc(DD*2);
  u16* waq = (u16*)alloc(DD*2); u16* wak = (u16*)alloc(DD*2); u16* wav = (u16*)alloc(DD*2);
  u16* wmp = (u16*)alloc(DD*2); u16* wo2 = (u16*)alloc(DD*2);
  u16* hid_bf = (u16*)alloc((size_t)NL*D*2);
  u16* pm_bf  = (u16*)alloc((size_t)NM*D*2);
  u16* qb = (u16*)alloc((size_t)NL*D*2);
  u16* kb = (u16*)alloc((size_t)NL*D*2);
  u16* vb = (u16*)alloc((size_t)NL*D*2);
  u16* ctx = (u16*)alloc((size_t)NL*D*2);
  float* tmpF  = (float*)alloc((size_t)NL*D*4);
  float* attnF = (float*)alloc((size_t)NL*D*4);
  u16* attnB = (u16*)alloc((size_t)NL*D*2);
  u16* interB = (u16*)alloc((size_t)NL*D*2);
  u16* st_bf  = (u16*)alloc((size_t)NM*D*2);
  float* tmpM = (float*)alloc((size_t)NM*D*4);
  float* c_f  = (float*)alloc((size_t)NM*D*4);
  float* z_f  = (float*)alloc((size_t)NM*D*4);
  u16* concatB = (u16*)alloc((size_t)NC*D*2);
  u16* kma = (u16*)alloc((size_t)NC*D*2);
  u16* vma = (u16*)alloc((size_t)NC*D*2);
  u16* mactx = (u16*)alloc((size_t)NL*D*2);
  u16* maprj = (u16*)alloc((size_t)NL*D*2);
  float* mamask = (float*)alloc((size_t)NC*4);

  float* out_mem   = (float*)d_out;
  float* out_layer = (float*)d_out + (size_t)NM*D;

  auto conv = [&](const float* src, u16* dst, size_t n) {
    f32_to_bf16_k<<<1024, 256, 0, stream>>>(src, dst, (int)n);
  };
  conv(W_attq, wq, DD); conv(W_attk, wk, DD); conv(W_attv, wv, DD);
  conv(W_atto, wo, DD); conv(W_int,  wi, DD);
  conv(W_muq, wmq, DD); conv(W_muk, wmk, DD); conv(W_muv, wmv, DD);
  conv(W_mc, wmc, DD);  conv(W_sc, wsc, DD);
  conv(W_mz, wmz, DD);  conv(W_sz, wsz, DD);
  conv(W_maq, waq, DD); conv(W_mak, wak, DD); conv(W_mav, wav, DD);
  conv(W_mp, wmp, DD);  conv(W_out, wo2, DD);
  conv(hidden, hid_bf, (size_t)NL*D);
  conv(prev_m, pm_bf,  (size_t)NM*D);

#define GEMM(FLAGS, A, B, bias, extra, oF, oB, M_, N_, K_)                   \
  gemm_bf16<FLAGS><<<dim3((N_) / 64, (M_) / 64), 128, 0, stream>>>(          \
      A, B, bias, extra, oF, oB, M_, N_, K_)

  // --- 1) self attention ---------------------------------------------------
  GEMM(1 | 64, hid_bf, wq, b_attq, nullptr, nullptr, qb, NL, D, D);
  GEMM(1 | 64, hid_bf, wk, b_attk, nullptr, nullptr, kb, NL, D, D);
  GEMM(1 | 64, hid_bf, wv, b_attv, nullptr, nullptr, vb, NL, D, D);
  flash_attn<<<dim3(LSEQ / 16, NHEAD, NB), 32, 0, stream>>>(
      qb, kb, vb, amask, ctx, LSEQ, LSEQ, /*memlen=*/0, /*causal=*/1);
  GEMM(1 | 32, ctx, wo, b_atto, nullptr, tmpF, nullptr, NL, D, D);
  add_ln_k<<<NL, 256, 0, stream>>>(tmpF, hidden, ln1_g, ln1_b, attnF, attnB);

  // --- 2) intermediate (gelu) ---------------------------------------------
  GEMM(1 | 2 | 64, attnB, wi, b_int, nullptr, nullptr, interB, NL, D, D);

  // --- 3) memory update ----------------------------------------------------
  GEMM(1 | 64, pm_bf,  wmq, b_muq, nullptr, nullptr, qb, NM, D, D);
  GEMM(1 | 64, interB, wmk, b_muk, nullptr, nullptr, kb, NL, D, D);
  GEMM(1 | 64, interB, wmv, b_muv, nullptr, nullptr, vb, NL, D, D);
  flash_attn<<<dim3(MMEM / 16, NHEAD, NB), 32, 0, stream>>>(
      qb, kb, vb, amask, st_bf, MMEM, LSEQ, /*memlen=*/LSEQ, /*causal=*/0);
  GEMM(32, pm_bf, wmc, nullptr, nullptr, tmpM, nullptr, NM, D, D);
  GEMM(1 | 4 | 16 | 32, st_bf, wsc, b_sc, tmpM, c_f, nullptr, NM, D, D);
  GEMM(32, pm_bf, wmz, nullptr, nullptr, tmpM, nullptr, NM, D, D);
  GEMM(1 | 8 | 16 | 32, st_bf, wsz, b_sz, tmpM, z_f, nullptr, NM, D, D);
  gate_update_k<<<2048, 256, 0, stream>>>(c_f, z_f, prev_m, out_mem, NM * D);

  // --- 4) memory-augmented attention --------------------------------------
  GEMM(1 | 64, interB, waq, b_maq, nullptr, nullptr, ctx, NL, D, D);
  concat_rows_k<<<4096, 256, 0, stream>>>(pm_bf, interB, concatB, NC * D);
  GEMM(1 | 64, concatB, wak, b_mak, nullptr, nullptr, kma, NC, D, D);
  GEMM(1 | 64, concatB, wav, b_mav, nullptr, nullptr, vma, NC, D, D);
  build_ma_mask_k<<<32, 256, 0, stream>>>(amask, mamask, NC);
  flash_attn<<<dim3(LSEQ / 16, NHEAD, NB), 32, 0, stream>>>(
      ctx, kma, vma, mamask, mactx, LSEQ, MMEM + LSEQ, /*memlen=*/MMEM,
      /*causal=*/1);
  GEMM(64, mactx, wmp, nullptr, nullptr, nullptr, maprj, NL, D, D);

  // --- 5) output dense + residual + LN ------------------------------------
  GEMM(1 | 32, maprj, wo2, b_out, nullptr, tmpF, nullptr, NL, D, D);
  add_ln_k<<<NL, 256, 0, stream>>>(tmpF, attnF, ln2_g, ln2_b, out_layer,
                                   nullptr);
#undef GEMM
}
